// TransformerLayer_65171833750118
// MI455X (gfx1250) — matrix operand order repaired and checked by reading, not compiled, not run on hardware
//
#include <hip/hip_runtime.h>
#include <math.h>

typedef __bf16 bf16_t;
typedef __attribute__((ext_vector_type(16))) __bf16 v16bf;
typedef __attribute__((ext_vector_type(8)))  __bf16 v8bf;
typedef __attribute__((ext_vector_type(8)))  float  v8f;

#define SEQ    2048
#define DMODEL 512
#define NHEAD  8
#define HDEPTH 64
#define MROWS  8192   // B * S

// ---------------------------------------------------------------- CDNA5 helpers
__device__ __forceinline__ v8f wmma_bf16(v16bf a, v16bf b, v8f c) {
  // (neg_a, A, neg_b, B, c_mod, C, reuse_a, reuse_b)
  return __builtin_amdgcn_wmma_f32_16x16x32_bf16(false, a, false, b, (short)0, c,
                                                 false, false);
}

// Generic pointer to a __shared__ object: low 32 bits are the LDS byte offset.
__device__ __forceinline__ unsigned lds_off(const void* p) {
  return (unsigned)(unsigned long long)p;
}

// Async copy 16B global -> LDS (per lane). ASYNCcnt-tracked Tensor/async path.
__device__ __forceinline__ void async_cp_b128(unsigned ldsByteOff, const void* g) {
  asm volatile("global_load_async_to_lds_b128 %0, %1, off"
               :: "v"(ldsByteOff), "v"((unsigned long long)g) : "memory");
}

__device__ __forceinline__ void wait_async0() {
#if __has_builtin(__builtin_amdgcn_s_wait_asynccnt)
  __builtin_amdgcn_s_wait_asynccnt(0);
#else
  asm volatile("s_wait_asynccnt 0x0" ::: "memory");
#endif
}

// A fragment: 16(M) x 32(K) bf16 from row-major tile, stride lda (elements).
__device__ __forceinline__ v16bf load_a_frag(const bf16_t* A, int lda) {
  const int lane = threadIdx.x & 31;
  const int m  = lane & 15;
  const int kh = lane >> 4;
  const bf16_t* p = A + m * lda + kh * 8;
  v8bf lo = *reinterpret_cast<const v8bf*>(p);        // K = kh*8 + 0..7
  v8bf hi = *reinterpret_cast<const v8bf*>(p + 16);   // K = 16 + kh*8 + 0..7
  v16bf f;
#pragma unroll
  for (int i = 0; i < 8; ++i) { f[i] = lo[i]; f[8 + i] = hi[i]; }
  return f;
}

// B fragment: 32(K) x 16(N) bf16 from Bt = B^T row-major (N x K), stride ldb.
__device__ __forceinline__ v16bf load_bt_frag(const bf16_t* Bt, int ldb) {
  const int lane = threadIdx.x & 31;
  const int n  = lane & 15;
  const int kh = lane >> 4;
  const bf16_t* p = Bt + n * ldb + kh * 8;
  v8bf lo = *reinterpret_cast<const v8bf*>(p);
  v8bf hi = *reinterpret_cast<const v8bf*>(p + 16);
  v16bf f;
#pragma unroll
  for (int i = 0; i < 8; ++i) { f[i] = lo[i]; f[8 + i] = hi[i]; }
  return f;
}

// ---------------------------------------------------------------- prep kernels
__global__ __launch_bounds__(256)
void cvt_bf16_kernel(const float* __restrict__ x, bf16_t* __restrict__ y, int n) {
  int i = blockIdx.x * 256 + threadIdx.x;
  if (i < n) y[i] = (bf16_t)x[i];
}

__global__ __launch_bounds__(256)
void wtrans_kernel(const float* __restrict__ w, bf16_t* __restrict__ wt) {
  int i = blockIdx.x * 256 + threadIdx.x;        // i = k*512 + n
  int k = i >> 9, n = i & 511;
  wt[n * DMODEL + k] = (bf16_t)w[i];
}

// ---------------------------------------------------------------- GEMM 8192x512x512
// Async double-buffered LDS staging of the 128x32 B tile (shared by 8 waves);
// A fragments straight from global. Epilogue fully compile-time specialized.
// OMODE: 0 row-major; 1 Q/K head layout [b][h][s][d]; 2 V transposed [b][h][d][s].
template <bool BIAS, bool RES, bool LEAKY, int OMODE, bool OUTF, bool OUTB>
__global__ __launch_bounds__(256)
void gemm512_kernel(const bf16_t* __restrict__ A, const bf16_t* __restrict__ Bt,
                    const float* __restrict__ bias, const float* __restrict__ residual,
                    float* __restrict__ outF, bf16_t* __restrict__ outB) {
  __shared__ bf16_t bsm[2][128 * 32];            // 2 x 8KB double buffer

  const int K = DMODEL, N = DMODEL;
  const int t    = threadIdx.x;
  const int w    = t >> 5;
  const int lane = t & 31;
  const int kh   = lane >> 4;
  const int nn   = lane & 15;
  const int m0    = blockIdx.x * 64 + (w & 3) * 16;
  const int n0blk = blockIdx.y * 128;
  const int nw    = (w >> 2) * 64;               // wave's N offset inside block tile

  // copy assignment: 128 rows x 64B, 4 x 16B chunks/row, 2 per thread
  const int r0 = t >> 2, c0 = t & 3;             // rows 0..63
  const unsigned lb0 = lds_off(&bsm[0][0]);
  const unsigned lb1 = lds_off(&bsm[1][0]);

  auto issue = [&](int kk, int buf) {
    unsigned lb = buf ? lb1 : lb0;
    const bf16_t* g0 = Bt + (size_t)(n0blk + r0) * K + kk + c0 * 8;
    const bf16_t* g1 = Bt + (size_t)(n0blk + r0 + 64) * K + kk + c0 * 8;
    async_cp_b128(lb + r0 * 64 + c0 * 16, g0);
    async_cp_b128(lb + (r0 + 64) * 64 + c0 * 16, g1);
  };

  v8f acc[4];
#pragma unroll
  for (int j = 0; j < 4; ++j)
#pragma unroll
    for (int r = 0; r < 8; ++r) acc[j][r] = 0.f;

  issue(0, 0);
  wait_async0();
  __syncthreads();

  const int NSTEP = K / 32;
  for (int s = 0; s < NSTEP; ++s) {
    const int kk  = s * 32;
    const int cur = s & 1;
    if (s + 1 < NSTEP) issue(kk + 32, cur ^ 1);

    __builtin_prefetch(A + (size_t)m0 * K + kk + 128, 0, 1);
    v16bf a = load_a_frag(A + (size_t)m0 * K + kk, K);
#pragma unroll
    for (int j = 0; j < 4; ++j) {
      v16bf b = load_bt_frag(&bsm[cur][(nw + 16 * j) * 32], 32);
      acc[j] = wmma_bf16(a, b, acc[j]);
    }
    __syncthreads();      // all waves done reading bsm[cur]
    wait_async0();        // own async copies for next step landed
    __syncthreads();      // everyone's copies landed
  }

#pragma unroll
  for (int j = 0; j < 4; ++j) {
#pragma unroll
    for (int r = 0; r < 8; ++r) {
      int m = m0 + r + 8 * kh;
      int n = n0blk + nw + 16 * j + nn;
      float v = acc[j][r];
      if (BIAS)  v += bias[n];
      if (RES)   v += residual[m * N + n];
      if (LEAKY) v = (v > 0.f) ? v : 0.2f * v;
      if (OMODE == 0) {
        if (OUTF) outF[m * N + n] = v;
        if (OUTB) outB[m * N + n] = (bf16_t)v;
      } else {
        int bb = m >> 11, ss = m & (SEQ - 1);
        int h  = n >> 6,  d  = n & (HDEPTH - 1);
        int idx = (OMODE == 1) ? ((bb * NHEAD + h) * SEQ + ss) * HDEPTH + d
                               : ((bb * NHEAD + h) * HDEPTH + d) * SEQ + ss;
        outB[idx] = (bf16_t)v;
      }
    }
  }
}

// ---------------------------------------------------------------- flash attention
// Qh,Kh: [b][h][s][64] bf16 ; Vt: [b][h][64][s] bf16 ; mask: [b][s] f32.
// 128 thr (4 waves); each wave owns a 16-row q tile; K/V chunks async-staged in
// LDS (double buffered) and shared by all 4 waves.
__global__ __launch_bounds__(128)
void attn_kernel(const bf16_t* __restrict__ Qh, const bf16_t* __restrict__ Kh,
                 const bf16_t* __restrict__ Vt, const float* __restrict__ mask,
                 bf16_t* __restrict__ ctx) {
  __shared__ bf16_t ksm[2][32 * 64];   // 2 x 4KB  (keys x depth)
  __shared__ bf16_t vsm[2][64 * 32];   // 2 x 4KB  (depth x keys)
  __shared__ float  pbuf[4 * 16 * 32]; // 8KB P relayout

  const int t    = threadIdx.x;
  const int w    = t >> 5;
  const int lane = t & 31;
  const int kh   = lane >> 4;
  const int nn   = lane & 15;
  const int bh = blockIdx.x >> 5;
  const int qt = blockIdx.x & 31;
  const int b  = bh >> 3;
  const int h  = bh & 7;
  const int q0 = qt * 64 + w * 16;

  const bf16_t* Qb = Qh + bh * SEQ * HDEPTH;
  const bf16_t* Kb = Kh + bh * SEQ * HDEPTH;
  const bf16_t* Vb = Vt + bh * HDEPTH * SEQ;
  float* pl = pbuf + w * 512;

  const unsigned lk0 = lds_off(&ksm[0][0]), lk1 = lds_off(&ksm[1][0]);
  const unsigned lv0 = lds_off(&vsm[0][0]), lv1 = lds_off(&vsm[1][0]);

  // K tile: 32 rows x 128B (8 chunks) = 256 assignments -> 2/thread
  // V tile: 64 rows x 64B  (4 chunks) = 256 assignments -> 2/thread
  auto issue = [&](int kc, int buf) {
    unsigned lk = buf ? lk1 : lk0;
    unsigned lv = buf ? lv1 : lv0;
#pragma unroll
    for (int a = 0; a < 2; ++a) {
      int id = t + a * 128;
      int kr = id >> 3, Kc = id & 7;             // key row, 16B chunk
      async_cp_b128(lk + kr * 128 + Kc * 16, Kb + (kc + kr) * HDEPTH + Kc * 8);
      int vr = id >> 2, Vc = id & 3;             // depth row, 16B chunk
      async_cp_b128(lv + vr * 64 + Vc * 16, Vb + vr * SEQ + kc + Vc * 8);
    }
  };

  const v16bf aq0 = load_a_frag(Qb + q0 * HDEPTH +  0, HDEPTH);
  const v16bf aq1 = load_a_frag(Qb + q0 * HDEPTH + 32, HDEPTH);

  float rmax[8], rsum[8];
  v8f cacc[4];
#pragma unroll
  for (int r = 0; r < 8; ++r) { rmax[r] = -3.0e38f; rsum[r] = 0.f; }
#pragma unroll
  for (int dn = 0; dn < 4; ++dn)
#pragma unroll
    for (int r = 0; r < 8; ++r) cacc[dn][r] = 0.f;

  issue(0, 0);
  wait_async0();
  __syncthreads();

  const int NC = SEQ / 32;
  for (int i = 0; i < NC; ++i) {
    const int kc  = i * 32;
    const int cur = i & 1;
    if (i + 1 < NC) issue(kc + 32, cur ^ 1);

    // ---- logits 16x32: 2 N-tiles x 2 K-steps of depth
    v8f lg[2];
#pragma unroll
    for (int j = 0; j < 2; ++j) {
#pragma unroll
      for (int r = 0; r < 8; ++r) lg[j][r] = 0.f;
      lg[j] = wmma_bf16(aq0, load_bt_frag(&ksm[cur][16 * j * 64 +  0], HDEPTH), lg[j]);
      lg[j] = wmma_bf16(aq1, load_bt_frag(&ksm[cur][16 * j * 64 + 32], HDEPTH), lg[j]);
    }
    // ---- scale + additive mask
#pragma unroll
    for (int j = 0; j < 2; ++j) {
      float mk = mask[b * SEQ + kc + 16 * j + nn] * -1.0e9f;
#pragma unroll
      for (int r = 0; r < 8; ++r) lg[j][r] = lg[j][r] * 0.125f + mk;
    }
    // ---- online softmax (row stats reduce across the owning 16-lane half)
    float cmax[8], corr[8], csum[8];
#pragma unroll
    for (int r = 0; r < 8; ++r) cmax[r] = fmaxf(lg[0][r], lg[1][r]);
#pragma unroll
    for (int xm = 1; xm <= 8; xm <<= 1)
#pragma unroll
      for (int r = 0; r < 8; ++r) cmax[r] = fmaxf(cmax[r], __shfl_xor(cmax[r], xm, 32));
#pragma unroll
    for (int r = 0; r < 8; ++r) {
      float nm = fmaxf(rmax[r], cmax[r]);
      corr[r]  = __expf(rmax[r] - nm);
      rmax[r]  = nm;
      csum[r]  = 0.f;
    }
#pragma unroll
    for (int j = 0; j < 2; ++j)
#pragma unroll
      for (int r = 0; r < 8; ++r) {
        float p = __expf(lg[j][r] - rmax[r]);
        lg[j][r] = p;
        csum[r] += p;
      }
#pragma unroll
    for (int xm = 1; xm <= 8; xm <<= 1)
#pragma unroll
      for (int r = 0; r < 8; ++r) csum[r] += __shfl_xor(csum[r], xm, 32);
#pragma unroll
    for (int r = 0; r < 8; ++r) rsum[r] = rsum[r] * corr[r] + csum[r];
#pragma unroll
    for (int dn = 0; dn < 4; ++dn)
#pragma unroll
      for (int r = 0; r < 8; ++r) cacc[dn][r] *= corr[r];

    // ---- P (C-frag) -> A-frag relayout via LDS, convert to bf16
#pragma unroll
    for (int j = 0; j < 2; ++j)
#pragma unroll
      for (int r = 0; r < 8; ++r)
        pl[(r + 8 * kh) * 32 + 16 * j + nn] = lg[j][r];
    __syncthreads();
    v16bf ap;
#pragma unroll
    for (int i2 = 0; i2 < 8; ++i2) {
      ap[i2]     = (bf16_t)pl[nn * 32 + kh * 8 + i2];
      ap[8 + i2] = (bf16_t)pl[nn * 32 + 16 + kh * 8 + i2];
    }

    // ---- ctx += P(16x32) @ V(32x64)
#pragma unroll
    for (int dn = 0; dn < 4; ++dn)
      cacc[dn] = wmma_bf16(ap, load_bt_frag(&vsm[cur][dn * 16 * 32], 32), cacc[dn]);

    __syncthreads();      // all reads of ksm/vsm[cur] + pbuf done
    wait_async0();        // own async copies for next chunk landed
    __syncthreads();      // everyone's copies landed
  }

  // ---- normalize and store ctx (head-concat row-major for the Wo GEMM)
#pragma unroll
  for (int r = 0; r < 8; ++r) {
    float inv = 1.0f / rsum[r];
    int s  = q0 + r + 8 * kh;
    int mg = b * SEQ + s;
#pragma unroll
    for (int dn = 0; dn < 4; ++dn)
      ctx[mg * DMODEL + h * HDEPTH + dn * 16 + nn] = (bf16_t)(cacc[dn][r] * inv);
  }
}

// ---------------------------------------------------------------- LayerNorm
__global__ __launch_bounds__(256)
void ln_kernel(const float* __restrict__ x, const float* __restrict__ g,
               const float* __restrict__ beta, float* __restrict__ outF,
               bf16_t* __restrict__ outB) {
  const int w    = threadIdx.x >> 5;
  const int lane = threadIdx.x & 31;
  const int row  = blockIdx.x * 8 + w;
  const float* xr = x + row * DMODEL;
  float v[16], s = 0.f, s2 = 0.f;
#pragma unroll
  for (int i = 0; i < 16; ++i) {
    v[i] = xr[lane + i * 32];
    s  += v[i];
    s2 += v[i] * v[i];
  }
#pragma unroll
  for (int xm = 1; xm <= 16; xm <<= 1) {
    s  += __shfl_xor(s,  xm, 32);
    s2 += __shfl_xor(s2, xm, 32);
  }
  float mean = s * (1.0f / DMODEL);
  float var  = s2 * (1.0f / DMODEL) - mean * mean;
  float inv  = rsqrtf(var + 1e-6f);
#pragma unroll
  for (int i = 0; i < 16; ++i) {
    int c = lane + i * 32;
    float y = (v[i] - mean) * inv * g[c] + beta[c];
    if (outF) outF[row * DMODEL + c] = y;
    if (outB) outB[row * DMODEL + c] = (bf16_t)y;
  }
}

// ---------------------------------------------------------------- launcher
extern "C" void kernel_launch(void* const* d_in, const int* in_sizes, int n_in,
                              void* d_out, int out_size, void* d_ws, size_t ws_size,
                              hipStream_t stream) {
  const float* q    = (const float*)d_in[0];
  const float* k    = (const float*)d_in[1];
  const float* mask = (const float*)d_in[2];
  const float* wq   = (const float*)d_in[3];
  const float* wk   = (const float*)d_in[4];
  const float* wv   = (const float*)d_in[5];
  const float* wo   = (const float*)d_in[6];
  const float* w1   = (const float*)d_in[7];
  const float* b1   = (const float*)d_in[8];
  const float* w2   = (const float*)d_in[9];
  const float* b2   = (const float*)d_in[10];
  const float* g1   = (const float*)d_in[11];
  const float* be1  = (const float*)d_in[12];
  const float* g2   = (const float*)d_in[13];
  const float* be2  = (const float*)d_in[14];

  char* base = (char*)d_ws;
  size_t off = 0;
  auto alloc = [&](size_t bytes) -> void* {
    void* p = base + off;
    off += (bytes + 255) & ~(size_t)255;
    return p;
  };
  const size_t actB = (size_t)MROWS * DMODEL;
  bf16_t* q_bf   = (bf16_t*)alloc(actB * 2);
  bf16_t* k_bf   = (bf16_t*)alloc(actB * 2);
  bf16_t* wqt    = (bf16_t*)alloc(DMODEL * DMODEL * 2);
  bf16_t* wkt    = (bf16_t*)alloc(DMODEL * DMODEL * 2);
  bf16_t* wvt    = (bf16_t*)alloc(DMODEL * DMODEL * 2);
  bf16_t* wot    = (bf16_t*)alloc(DMODEL * DMODEL * 2);
  bf16_t* w1t    = (bf16_t*)alloc(DMODEL * DMODEL * 2);
  bf16_t* w2t    = (bf16_t*)alloc(DMODEL * DMODEL * 2);
  bf16_t* Qh     = (bf16_t*)alloc(actB * 2);
  bf16_t* Kh     = (bf16_t*)alloc(actB * 2);
  bf16_t* Vt     = (bf16_t*)alloc(actB * 2);
  bf16_t* ctx_bf = (bf16_t*)alloc(actB * 2);
  float*  out1   = (float*) alloc(actB * 4);
  bf16_t* out1bf = (bf16_t*)alloc(actB * 2);
  bf16_t* h_bf   = (bf16_t*)alloc(actB * 2);
  float*  ypre   = (float*) alloc(actB * 4);

  const int nElem = (int)actB;
  cvt_bf16_kernel<<<nElem / 256, 256, 0, stream>>>(q, q_bf, nElem);
  cvt_bf16_kernel<<<nElem / 256, 256, 0, stream>>>(k, k_bf, nElem);
  wtrans_kernel<<<1024, 256, 0, stream>>>(wq, wqt);
  wtrans_kernel<<<1024, 256, 0, stream>>>(wk, wkt);
  wtrans_kernel<<<1024, 256, 0, stream>>>(wv, wvt);
  wtrans_kernel<<<1024, 256, 0, stream>>>(wo, wot);
  wtrans_kernel<<<1024, 256, 0, stream>>>(w1, w1t);
  wtrans_kernel<<<1024, 256, 0, stream>>>(w2, w2t);

  dim3 gg(MROWS / 64, DMODEL / 128);
  // QKV projections (head layouts)
  gemm512_kernel<false,false,false,1,false,true><<<gg, 256, 0, stream>>>(
      q_bf, wqt, nullptr, nullptr, nullptr, Qh);
  gemm512_kernel<false,false,false,1,false,true><<<gg, 256, 0, stream>>>(
      k_bf, wkt, nullptr, nullptr, nullptr, Kh);
  gemm512_kernel<false,false,false,2,false,true><<<gg, 256, 0, stream>>>(
      k_bf, wvt, nullptr, nullptr, nullptr, Vt);
  // flash attention
  attn_kernel<<<1024, 128, 0, stream>>>(Qh, Kh, Vt, mask, ctx_bf);
  // output projection + residual(q) -> LN1
  gemm512_kernel<false,true,false,0,true,false><<<gg, 256, 0, stream>>>(
      ctx_bf, wot, nullptr, q, out1, nullptr);
  ln_kernel<<<1024, 256, 0, stream>>>(out1, g1, be1, out1, out1bf);
  // FFN
  gemm512_kernel<true,false,true,0,false,true><<<gg, 256, 0, stream>>>(
      out1bf, w1t, b1, nullptr, nullptr, h_bf);
  gemm512_kernel<true,true,false,0,true,false><<<gg, 256, 0, stream>>>(
      h_bf, w2t, b2, out1, ypre, nullptr);
  // LN2 -> final output (f32)
  ln_kernel<<<1024, 256, 0, stream>>>(ypre, g2, be2, (float*)d_out, nullptr);
}
